// RVQAudioQuantizer_72499047957279
// MI455X (gfx1250) — matrix operand order repaired and checked by reading, not compiled, not run on hardware
//
#include <hip/hip_runtime.h>
#include <hip/hip_bf16.h>

// ---------------------------------------------------------------------------
// RVQ audio quantizer for gfx1250 (MI455X).
//   feats:     [B=16, T=1024, D=1280]  f32
//   lengths:   [B]                     i64
//   codebooks: [Q=4, K=256, D=1280]    f32
// Outputs (flat in d_out, f32): quantized_out [B,T,D], indices [B,T,Q] (as
// float, -1 for masked), total_loss (1).
//
// Strategy: residual tile stays resident in fp32 LDS across all 4 stages
// (feats read once from HBM); per-stage 16384x1280x256 dots computed with
// v_wmma_f32_16x16x32_bf16. |r|^2 cancels out of the argmin, so distance
// reduces to |c|^2 - 2*dot. K-outer / N-inner loop with 4 accumulators so
// each A fragment (LDS->bf16 conversion) feeds 4 WMMAs.
// ---------------------------------------------------------------------------

typedef __attribute__((ext_vector_type(16))) __bf16 v16bf;
typedef __attribute__((ext_vector_type(8)))  float  v8f;
typedef __attribute__((ext_vector_type(4)))  float  v4f;

static constexpr int NUM_Q  = 4;
static constexpr int CB_K   = 256;
static constexpr int DIM    = 1280;
static constexpr int B_SZ   = 16;
static constexpr int T_SZ   = 1024;
static constexpr int M_TOT  = B_SZ * T_SZ;        // 16384 rows
static constexpr int ROWS   = 16;                 // rows per block
static constexpr int RF_STR = DIM + 4;            // fp32 LDS row stride (pad)
static constexpr size_t QOUT_OFF = (size_t)M_TOT * DIM;          // 20,971,520
static constexpr size_t IDX_OFF  = QOUT_OFF;                     // indices base
static constexpr size_t LOSS_OFF = QOUT_OFF + (size_t)M_TOT * NUM_Q;

// ---------------------------------------------------------------------------
// Prep: codebooks fp32 -> bf16 in workspace, plus |c|^2 per codebook entry.
// grid = NUM_Q*CB_K blocks, 256 threads.
// ---------------------------------------------------------------------------
__global__ void rvq_prep_kernel(const float* __restrict__ cb,
                                __bf16* __restrict__ cb_bf,
                                float* __restrict__ c2) {
    const int row = blockIdx.x;                   // q*256 + n
    const int tid = threadIdx.x;
    const float* src = cb + (size_t)row * DIM;
    __bf16* dst = cb_bf + (size_t)row * DIM;
    float acc = 0.f;
    for (int i = tid; i < DIM; i += 256) {
        float v = src[i];
        dst[i] = (__bf16)v;
        acc += v * v;
    }
    __shared__ float red[256];
    red[tid] = acc;
    __syncthreads();
    for (int s = 128; s > 0; s >>= 1) {
        if (tid < s) red[tid] += red[tid + s];
        __syncthreads();
    }
    if (tid == 0) c2[row] = red[0];
}

// ---------------------------------------------------------------------------
// Main kernel: 1024 blocks x 128 threads (4 waves). Block owns 16 rows.
// Wave w computes dot columns [w*64, w*64+64) as 4 WMMA N-tiles of 16,
// accumulated simultaneously so every A fragment feeds 4 WMMAs.
// ---------------------------------------------------------------------------
struct SMem {
    float  rf[ROWS][RF_STR];       // fp32 residual tile (master copy)
    float  best[4][ROWS];          // per-wave best score per row
    int    bidx[4][ROWS];          // per-wave best index per row
    int    fidx[ROWS];             // final argmin per row (this stage)
    int    valid[ROWS];            // mask per row (t < length[b])
    float  red[128];               // loss reduction
};

__global__ __launch_bounds__(128)
void rvq_main_kernel(const float* __restrict__ feats,
                     const long long* __restrict__ lengths,
                     const float* __restrict__ codebooks,
                     const __bf16* __restrict__ cb_bf,
                     const float* __restrict__ c2,
                     float* __restrict__ out) {
    __shared__ SMem sm;

    const int tid  = threadIdx.x;
    const int wave = tid >> 5;
    const int lane = tid & 31;
    const int mg0  = blockIdx.x * ROWS;

    // ---- load residual tile (fp32) and row validity ----
    for (int row = 0; row < ROWS; ++row) {
        const size_t gbase = (size_t)(mg0 + row) * DIM;
        #pragma unroll
        for (int j = 0; j < DIM / 128; ++j) {
            const int d = tid + 128 * j;
            sm.rf[row][d] = feats[gbase + d];
        }
    }
    if (tid < ROWS) {
        const int mg = mg0 + tid;
        const int b  = mg >> 10;            // / T
        const int t  = mg & (T_SZ - 1);     // % T
        sm.valid[tid] = (t < (int)lengths[b]) ? 1 : 0;
    }

    float loss_acc = 0.f;

    const int m    = lane & 15;             // A row this lane supplies
    const int koff = (lane >= 16) ? 16 : 0; // A-frag K split (contiguous 16-K halves)
    const int kb2  = (lane >= 16) ? 16 : 0; // B-frag K split

    // =======================  4 RVQ stages  =======================
    for (int q = 0; q < NUM_Q; ++q) {
        __syncthreads();   // residual tile (and valid[]) ready

        // ---- dots via WMMA: 4 N-tiles accumulated together ----
        // per-lane column for N-tile nt: n(nt) = (wave*4+nt)*16 + (lane&15)
        const __bf16* bp[4];
        #pragma unroll
        for (int nt = 0; nt < 4; ++nt) {
            const int n = (wave * 4 + nt) * 16 + (lane & 15);
            bp[nt] = cb_bf + ((size_t)(q * CB_K + n)) * DIM + kb2;
        }

        v8f acc0 = {0.f,0.f,0.f,0.f,0.f,0.f,0.f,0.f};
        v8f acc1 = acc0, acc2 = acc0, acc3 = acc0;

        for (int kb = 0; kb < DIM; kb += 32) {
            // A fragment: 16x32 bf16, built once from fp32 LDS, used 4x.
            // lanes 0-15:  halves 0-15 = K kb..kb+15
            // lanes 16-31: halves 0-15 = K kb+16..kb+31
            const float* ar = &sm.rf[m][kb + koff];
            v4f f0 = *(const v4f*)(ar);
            v4f f1 = *(const v4f*)(ar + 4);
            v4f f2 = *(const v4f*)(ar + 8);
            v4f f3 = *(const v4f*)(ar + 12);
            v16bf a;
            #pragma unroll
            for (int i = 0; i < 4; ++i) {
                a[i]      = (__bf16)f0[i];
                a[4 + i]  = (__bf16)f1[i];
                a[8 + i]  = (__bf16)f2[i];
                a[12 + i] = (__bf16)f3[i];
            }
            // B fragments: 32x16 bf16; column n, 16 contiguous K values.
            v16bf b0 = *(const v16bf*)(bp[0] + kb);
            v16bf b1 = *(const v16bf*)(bp[1] + kb);
            v16bf b2 = *(const v16bf*)(bp[2] + kb);
            v16bf b3 = *(const v16bf*)(bp[3] + kb);

            acc0 = __builtin_amdgcn_wmma_f32_16x16x32_bf16(
                false, a, false, b0, (short)0, acc0, false, false);
            acc1 = __builtin_amdgcn_wmma_f32_16x16x32_bf16(
                false, a, false, b1, (short)0, acc1, false, false);
            acc2 = __builtin_amdgcn_wmma_f32_16x16x32_bf16(
                false, a, false, b2, (short)0, acc2, false, false);
            acc3 = __builtin_amdgcn_wmma_f32_16x16x32_bf16(
                false, a, false, b3, (short)0, acc3, false, false);
        }

        // ---- score = |c|^2 - 2*dot ; per-lane argmin over 4 N-tiles ----
        float bestv[8];
        int   besti[8];
        {
            const v8f accs[4] = {acc0, acc1, acc2, acc3};
            #pragma unroll
            for (int v = 0; v < 8; ++v) { bestv[v] = 3.4e38f; besti[v] = 0x7fffffff; }
            #pragma unroll
            for (int nt = 0; nt < 4; ++nt) {
                const int n = (wave * 4 + nt) * 16 + (lane & 15);
                const float c2n = c2[q * CB_K + n];
                #pragma unroll
                for (int v = 0; v < 8; ++v) {
                    const float score = __builtin_fmaf(-2.0f, accs[nt][v], c2n);
                    if (score < bestv[v]) { bestv[v] = score; besti[v] = n; }
                }
            }
        }

        // ---- cross-lane argmin within each 16-lane half (C layout:
        //      lanes 0-15 hold rows 0-7, lanes 16-31 hold rows 8-15) ----
        #pragma unroll
        for (int v = 0; v < 8; ++v) {
            #pragma unroll
            for (int off = 1; off < 16; off <<= 1) {
                const float ov = __shfl_xor(bestv[v], off, 32);
                const int   oi = __shfl_xor(besti[v], off, 32);
                if (ov < bestv[v] || (ov == bestv[v] && oi < besti[v])) {
                    bestv[v] = ov; besti[v] = oi;
                }
            }
        }
        if (lane == 0 || lane == 16) {
            const int rbase = (lane == 0) ? 0 : 8;
            #pragma unroll
            for (int v = 0; v < 8; ++v) {
                sm.best[wave][rbase + v] = bestv[v];
                sm.bidx[wave][rbase + v] = besti[v];
            }
        }
        __syncthreads();

        // ---- cross-wave argmin (first-occurrence tie-break) + idx output ----
        if (tid < ROWS) {
            const int row = tid;
            float bv = sm.best[0][row];
            int   bi = sm.bidx[0][row];
            #pragma unroll
            for (int w = 1; w < 4; ++w) {
                const float ov = sm.best[w][row];
                const int   oi = sm.bidx[w][row];
                if (ov < bv || (ov == bv && oi < bi)) { bv = ov; bi = oi; }
            }
            sm.fidx[row] = bi;
            const int mg = mg0 + row;
            out[IDX_OFF + (size_t)mg * NUM_Q + q] =
                sm.valid[row] ? (float)bi : -1.0f;
        }
        __syncthreads();

        // ---- gather + residual update (fp32) + masked commitment loss ----
        for (int row = 0; row < ROWS; ++row) {
            const int ci = sm.fidx[row];
            const float* qrow = codebooks + ((size_t)q * CB_K + ci) * DIM;
            const bool vld = sm.valid[row] != 0;
            #pragma unroll
            for (int j = 0; j < DIM / 128; ++j) {
                const int d = tid + 128 * j;
                const float diff = sm.rf[row][d] - qrow[d];
                sm.rf[row][d] = diff;              // residual -= quantized
                if (vld) loss_acc = __builtin_fmaf(diff, diff, loss_acc);
            }
        }
    }
    __syncthreads();

    // ---- quantized_out = x - residual_final (masked rows pass through x) ----
    for (int row = 0; row < ROWS; ++row) {
        const int mg = mg0 + row;
        const size_t gbase = (size_t)mg * DIM;
        const bool vld = sm.valid[row] != 0;
        #pragma unroll
        for (int j = 0; j < DIM / 128; ++j) {
            const int d = tid + 128 * j;
            const float x = feats[gbase + d];
            out[gbase + d] = vld ? (x - sm.rf[row][d]) : x;
        }
    }

    // ---- loss reduction + global accumulate ----
    sm.red[tid] = loss_acc;
    __syncthreads();
    for (int s = 64; s > 0; s >>= 1) {
        if (tid < s) sm.red[tid] += sm.red[tid + s];
        __syncthreads();
    }
    if (tid == 0) {
        long long sl = 0;
        for (int b = 0; b < B_SZ; ++b) sl += lengths[b];
        const float nv = fmaxf((float)sl * (float)DIM, 1.0f);
        atomicAdd(&out[LOSS_OFF], sm.red[0] / nv);
    }
}

// ---------------------------------------------------------------------------
extern "C" void kernel_launch(void* const* d_in, const int* in_sizes, int n_in,
                              void* d_out, int out_size, void* d_ws, size_t ws_size,
                              hipStream_t stream) {
    (void)in_sizes; (void)n_in; (void)out_size; (void)ws_size;

    const float*     feats     = (const float*)d_in[0];
    const long long* lengths   = (const long long*)d_in[1];
    const float*     codebooks = (const float*)d_in[2];
    float*           out       = (float*)d_out;

    // Workspace layout: bf16 codebooks (2.56 MB) then |c|^2 (4 KB).
    __bf16* cb_bf = (__bf16*)d_ws;
    float*  c2    = (float*)((char*)d_ws +
                             (size_t)NUM_Q * CB_K * DIM * sizeof(__bf16));

    rvq_prep_kernel<<<NUM_Q * CB_K, 256, 0, stream>>>(codebooks, cb_bf, c2);
    hipMemsetAsync(out + LOSS_OFF, 0, sizeof(float), stream);
    rvq_main_kernel<<<M_TOT / ROWS, 128, 0, stream>>>(
        feats, lengths, codebooks, cb_bf, c2, out);
}
